// RGCNRepresentations_79044578116389
// MI455X (gfx1250) — matrix operand order repaired and checked by reading, not compiled, not run on hardware
//
#include <hip/hip_runtime.h>

#define DIM 128
#define TEDGE 64   // edges per block in msg kernel (4 sub-tiles of 16)

typedef __attribute__((ext_vector_type(16))) __bf16        v16bf;
typedef __attribute__((ext_vector_type(8)))  float         v8f;
typedef __attribute__((ext_vector_type(16))) unsigned short v16u;
typedef __attribute__((ext_vector_type(8)))  unsigned short v8u;
typedef int v4i_vs __attribute__((vector_size(16)));   // matches builtin param type

#if defined(__has_builtin)
#if __has_builtin(__builtin_amdgcn_global_load_async_to_lds_b128) && \
    __has_builtin(__builtin_amdgcn_s_wait_asynccnt)
#define USE_ASYNC_LDS 1
#endif
#endif

#define AS1 __attribute__((address_space(1)))
#define AS3 __attribute__((address_space(3)))

__device__ __forceinline__ unsigned short f2bf(float f) {
    unsigned int u = __float_as_uint(f);
    unsigned int r = u + 0x7FFFu + ((u >> 16) & 1u);   // round-to-nearest-even
    return (unsigned short)(r >> 16);
}

// ---------------- pass 1: in-degree + relation histogram ----------------
__global__ void deg_hist_kernel(const int* __restrict__ tgt, const int* __restrict__ etype,
                                float* __restrict__ indeg, int* __restrict__ relcnt, int E) {
    int e = blockIdx.x * blockDim.x + threadIdx.x;
    if (e < E) {
        atomicAdd(&indeg[tgt[e]], 1.0f);
        atomicAdd(&relcnt[etype[e]], 1);
    }
}

// ---------------- tiny scan over R relations ----------------
__global__ void scan_kernel(const int* __restrict__ relcnt, int* __restrict__ edgeOff,
                            int* __restrict__ tileOff, int* __restrict__ totalTiles, int R) {
    if (blockIdx.x == 0 && threadIdx.x == 0) {
        int off = 0, toff = 0;
        for (int r = 0; r < R; ++r) {
            edgeOff[r] = off; tileOff[r] = toff;
            off  += relcnt[r];
            toff += (relcnt[r] + TEDGE - 1) / TEDGE;
        }
        edgeOff[R] = off; tileOff[R] = toff;
        *totalTiles = toff;
    }
}

// ---------------- counting-sort permutation by relation ----------------
__global__ void perm_kernel(const int* __restrict__ etype, const int* __restrict__ edgeOff,
                            int* __restrict__ cursor, int* __restrict__ perm, int E) {
    int e = blockIdx.x * blockDim.x + threadIdx.x;
    if (e < E) {
        int r = etype[e];
        int p = atomicAdd(&cursor[r], 1);
        perm[edgeOff[r] + p] = e;
    }
}

// ---------------- tile list: (relation, edge-start) per TEDGE-edge tile ----------------
__global__ void tile_kernel(const int* __restrict__ relcnt, const int* __restrict__ edgeOff,
                            const int* __restrict__ tileOff,
                            int* __restrict__ tile_rel, int* __restrict__ tile_start) {
    int r = blockIdx.x;
    int nt = (relcnt[r] + TEDGE - 1) / TEDGE;
    int tb = tileOff[r], eb = edgeOff[r];
    for (int i = threadIdx.x; i < nt; i += blockDim.x) {
        tile_rel[tb + i]   = r;
        tile_start[tb + i] = eb + i * TEDGE;
    }
}

// ---------------- W_r^T = (sum_b coef[r,b]*bases[b])^T in bf16  [r][n][k] ----------------
__global__ void make_w_kernel(const float* __restrict__ bases, const float* __restrict__ coefs_l,
                              unsigned short* __restrict__ Wt, int NB) {
    int r = blockIdx.x;
    float c[16];
    for (int b = 0; b < NB; ++b) c[b] = coefs_l[r * NB + b];
    unsigned short* out = Wt + r * (DIM * DIM);
    for (int idx = threadIdx.x; idx < DIM * DIM; idx += blockDim.x) {
        int n = idx >> 7, k = idx & 127;
        float s = 0.f;
        for (int b = 0; b < NB; ++b)
            s += c[b] * bases[b * (DIM * DIM) + k * DIM + n];
        out[idx] = f2bf(s);                 // Wt[r][n][k]
    }
}

// ---------------- self_w[l]^T -> bf16 [n][k] ----------------
__global__ void transpose_bf16_kernel(const float* __restrict__ w, unsigned short* __restrict__ wt) {
    int idx = blockIdx.x * blockDim.x + threadIdx.x;
    if (idx < DIM * DIM) {
        int n = idx >> 7, k = idx & 127;
        wt[idx] = f2bf(w[k * DIM + n]);
    }
}

// ---------------- x (f32) -> bf16 ----------------
__global__ void cvt_bf16_kernel(const float* __restrict__ x, unsigned short* __restrict__ xbf, int n) {
    int i = (blockIdx.x * blockDim.x + threadIdx.x) * 8;
    if (i >= n) return;
    int m = n - i; if (m > 8) m = 8;
    for (int j = 0; j < m; ++j) xbf[i + j] = f2bf(x[i + j]);
}

// ---- edge-message GEMM: 64 edges x 128 cols per block, B held in registers ----
__global__ void __launch_bounds__(256)
msg_kernel(const unsigned short* __restrict__ xbf,
           const unsigned short* __restrict__ Wt,
           const int* __restrict__ perm,
           const int* __restrict__ source, const int* __restrict__ target,
           const float* __restrict__ indeg,
           const int* __restrict__ edgeOff,
           const int* __restrict__ tile_rel, const int* __restrict__ tile_start,
           const int* __restrict__ totalTiles,
           float* __restrict__ agg) {
    int t = blockIdx.x;
    if (t >= *totalTiles) return;                 // uniform per block

    int r     = tile_rel[t];
    int start = tile_start[t];
    int cnt   = edgeOff[r + 1] - start;
    if (cnt > TEDGE) cnt = TEDGE;

    __shared__ __align__(32) unsigned short As[TEDGE * DIM];  // 16 KB
    __shared__ int   stgt[TEDGE];                 // pre-scaled: target*DIM
    __shared__ float sew[TEDGE];

    int tid = threadIdx.x;
    // stage A: TEDGE rows x 128 bf16; 1024 16-byte segments, 4 per thread
#pragma unroll
    for (int it = 0; it < (TEDGE * 16) / 256; ++it) {
        int p   = tid + 256 * it;
        int row = p >> 4;
        int seg = p & 15;
        if (row < cnt) {
            int e = perm[start + row];
            if (seg == 0) {
                int tg = target[e];
                stgt[row] = tg * DIM;
                sew[row]  = 1.0f / fmaxf(indeg[tg], 1.0f);
            }
            int soff = source[e] * DIM + seg * 8;
#ifdef USE_ASYNC_LDS
            __builtin_amdgcn_global_load_async_to_lds_b128(
                (AS1 v4i_vs*)(AS1 void*)(void*)(xbf + soff),
                (AS3 v4i_vs*)(AS3 void*)&As[row * DIM + seg * 8], 0, 0);
#else
            *(v8u*)&As[row * DIM + seg * 8] = *(const v8u*)(xbf + soff);
#endif
        } else {
            v8u z = {};
            *(v8u*)&As[row * DIM + seg * 8] = z;
            if (seg == 0) { stgt[row] = row * DIM; sew[row] = 0.f; }  // inert, spread
        }
    }
#ifdef USE_ASYNC_LDS
    __builtin_amdgcn_s_wait_asynccnt(0);
#endif
    __syncthreads();

    int wave = tid >> 5;
    int lane = tid & 31;
    int hf   = lane >> 4;        // lane half (ISA A/B/C layouts)
    int l16  = lane & 15;
    int n0   = wave * 16;        // 8 waves cover 128 output cols
    int col  = n0 + l16;

    // B fragments for this wave's 16-col strip, all K: 4 x v16u in registers
    const unsigned short* bcol = Wt + (r * DIM + col) * DIM;
    v16bf bfrag[4];
#pragma unroll
    for (int ki = 0; ki < 4; ++ki)
        bfrag[ki] = __builtin_bit_cast(v16bf, *(const v16u*)(bcol + 32 * ki + 16 * hf));

#pragma unroll
    for (int st = 0; st < TEDGE / 16; ++st) {
        const unsigned short* arow = &As[(st * 16 + l16) * DIM];
        v8f c = {};
#pragma unroll
        for (int ki = 0; ki < 4; ++ki) {
            int k0 = 32 * ki + 16 * hf;
            v8u alo = *(const v8u*)(arow + k0);
            v8u ahi = *(const v8u*)(arow + k0 + 8);
            v16u au = __builtin_shufflevector(alo, ahi, 0,1,2,3,4,5,6,7,8,9,10,11,12,13,14,15);
            v16bf a = __builtin_bit_cast(v16bf, au);
            c = __builtin_amdgcn_wmma_f32_16x16x32_bf16(false, a, false, bfrag[ki],
                                                        (short)0, c, false, false);
        }
        // straight-line scatter: padded rows have ew==0 -> adds 0.0f (inert)
#pragma unroll
        for (int i = 0; i < 8; ++i) {
            int m = st * 16 + i + 8 * hf;                  // C row per ISA layout
            atomicAdd(agg + (stgt[m] + col), sew[m] * c[i]);   // 32-bit offset
        }
    }
}

// ---- x_next = relu(agg + x @ self_w + bias): 64 rows per block, B in registers ----
__global__ void __launch_bounds__(256)
self_update_kernel(const unsigned short* __restrict__ xbf,
                   const unsigned short* __restrict__ swt,
                   const float* __restrict__ agg,
                   const float* __restrict__ bias_l,
                   float* __restrict__ out, int N) {
    int m0  = blockIdx.x * TEDGE;
    int tid = threadIdx.x;
    int wave = tid >> 5, lane = tid & 31, hf = lane >> 4, l16 = lane & 15;
    int n0 = wave * 16;
    int col = n0 + l16;

    const unsigned short* bcol = swt + col * DIM;
    v16bf bfrag[4];
#pragma unroll
    for (int ki = 0; ki < 4; ++ki)
        bfrag[ki] = __builtin_bit_cast(v16bf, *(const v16u*)(bcol + 32 * ki + 16 * hf));

    float bv = bias_l[col];

#pragma unroll
    for (int st = 0; st < TEDGE / 16; ++st) {
        int arow = m0 + st * 16 + l16;
        if (arow >= N) arow = N - 1;                        // clamp (pads harmless)
        const unsigned short* ap = xbf + arow * DIM;
        v8f c = {};
#pragma unroll
        for (int ki = 0; ki < 4; ++ki) {
            int k0 = 32 * ki + 16 * hf;
            v8u alo = *(const v8u*)(ap + k0);
            v8u ahi = *(const v8u*)(ap + k0 + 8);
            v16u au = __builtin_shufflevector(alo, ahi, 0,1,2,3,4,5,6,7,8,9,10,11,12,13,14,15);
            v16bf a = __builtin_bit_cast(v16bf, au);
            c = __builtin_amdgcn_wmma_f32_16x16x32_bf16(false, a, false, bfrag[ki],
                                                        (short)0, c, false, false);
        }
#pragma unroll
        for (int i = 0; i < 8; ++i) {
            int m = m0 + st * 16 + i + 8 * hf;
            if (m < N) {
                int off = m * DIM + col;                    // 32-bit offset
                float v = c[i] + agg[off] + bv;
                out[off] = fmaxf(v, 0.f);
            }
        }
    }
}

extern "C" void kernel_launch(void* const* d_in, const int* in_sizes, int n_in,
                              void* d_out, int out_size, void* d_ws, size_t ws_size,
                              hipStream_t stream) {
    const float* x0    = (const float*)d_in[0];
    const float* bases = (const float*)d_in[1];
    const float* coefs = (const float*)d_in[2];
    const float* selfw = (const float*)d_in[3];
    const float* bias  = (const float*)d_in[4];
    const int* source  = (const int*)d_in[5];
    const int* target  = (const int*)d_in[6];
    const int* etype   = (const int*)d_in[7];

    int N  = in_sizes[0] / DIM;
    int NB = in_sizes[1] / (DIM * DIM);
    int L  = in_sizes[3] / (DIM * DIM);
    int R  = in_sizes[2] / (L * NB);
    int E  = in_sizes[5];
    int maxTiles = (E + TEDGE - 1) / TEDGE + R;

    // workspace carve (256B aligned chunks)
    char* p = (char*)d_ws;
    auto carve = [&](size_t bytes) -> char* {
        char* q = p; p += (bytes + 255) & ~(size_t)255; return q;
    };
    float* indeg       = (float*)carve((size_t)N * 4);
    int*   relcnt      = (int*)carve((size_t)R * 4);
    int*   edgeOff     = (int*)carve((size_t)(R + 1) * 4);
    int*   tileOff     = (int*)carve((size_t)(R + 1) * 4);
    int*   cursor      = (int*)carve((size_t)R * 4);
    int*   totalTiles  = (int*)carve(4);
    int*   perm        = (int*)carve((size_t)E * 4);
    int*   tile_rel    = (int*)carve((size_t)maxTiles * 4);
    int*   tile_start  = (int*)carve((size_t)maxTiles * 4);
    unsigned short* Wt  = (unsigned short*)carve((size_t)R * DIM * DIM * 2);
    unsigned short* swt = (unsigned short*)carve((size_t)DIM * DIM * 2);
    unsigned short* xbf = (unsigned short*)carve((size_t)N * DIM * 2);
    float* agg = (float*)carve((size_t)N * DIM * 4);
    float* x1  = (float*)carve((size_t)N * DIM * 4);

    (void)hipMemsetAsync(indeg,  0, (size_t)N * 4, stream);
    (void)hipMemsetAsync(relcnt, 0, (size_t)R * 4, stream);
    (void)hipMemsetAsync(cursor, 0, (size_t)R * 4, stream);

    int eb = (E + 255) / 256;
    deg_hist_kernel<<<eb, 256, 0, stream>>>(target, etype, indeg, relcnt, E);
    scan_kernel<<<1, 32, 0, stream>>>(relcnt, edgeOff, tileOff, totalTiles, R);
    perm_kernel<<<eb, 256, 0, stream>>>(etype, edgeOff, cursor, perm, E);
    tile_kernel<<<R, 64, 0, stream>>>(relcnt, edgeOff, tileOff, tile_rel, tile_start);

    const float* xin = x0;
    for (int l = 0; l < L; ++l) {
        float* xout = (l == L - 1) ? (float*)d_out : x1;
        (void)hipMemsetAsync(agg, 0, (size_t)N * DIM * 4, stream);
        make_w_kernel<<<R, 256, 0, stream>>>(bases, coefs + (size_t)l * R * NB, Wt, NB);
        transpose_bf16_kernel<<<(DIM * DIM + 255) / 256, 256, 0, stream>>>(
            selfw + (size_t)l * DIM * DIM, swt);
        int ne = N * DIM;
        cvt_bf16_kernel<<<((ne + 7) / 8 + 255) / 256, 256, 0, stream>>>(xin, xbf, ne);
        msg_kernel<<<maxTiles, 256, 0, stream>>>(xbf, Wt, perm, source, target, indeg,
                                                 edgeOff, tile_rel, tile_start, totalTiles, agg);
        self_update_kernel<<<(N + TEDGE - 1) / TEDGE, 256, 0, stream>>>(
            xbf, swt, agg, bias + (size_t)l * DIM, xout, N);
        xin = xout;
    }
}
